// Attention_68504728371431
// MI455X (gfx1250) — matrix operand order repaired and checked by reading, not compiled, not run on hardware
//
#include <hip/hip_runtime.h>

#define BB 4
#define CC 256
#define CI 128
#define NN 4096                 // H*W
#define CN (CC * NN)            // 1048576 elements per batch of x / wy

typedef __attribute__((ext_vector_type(16))) __bf16 v16bf;
typedef __attribute__((ext_vector_type(8)))  float  v8f;
typedef __attribute__((ext_vector_type(8)))  unsigned short u16x8;
typedef __attribute__((ext_vector_type(16))) unsigned short u16x16;
typedef __attribute__((ext_vector_type(4)))  float  f32x4;

// fp32 -> bf16 round-to-nearest-even (bit trick; avoids relying on __bf16 arithmetic)
static __device__ __forceinline__ unsigned short f2bf(float f) {
  unsigned int u = __builtin_bit_cast(unsigned int, f);
  u += 0x7FFFu + ((u >> 16) & 1u);
  return (unsigned short)(u >> 16);
}

// pack two fp32 -> packed bf16x2 (for 32-bit LDS stores)
static __device__ __forceinline__ unsigned int f2bf2(float lo, float hi) {
  return (unsigned int)f2bf(lo) | ((unsigned int)f2bf(hi) << 16);
}

static __device__ __forceinline__ v16bf mkfrag(u16x8 lo, u16x8 hi) {
  u16x16 w = __builtin_shufflevector(lo, hi, 0, 1, 2, 3, 4, 5, 6, 7,
                                             8, 9, 10, 11, 12, 13, 14, 15);
  return __builtin_bit_cast(v16bf, w);
}

static __device__ __forceinline__ v8f wmma_bf16(v16bf a, v16bf b, v8f c) {
  return __builtin_amdgcn_wmma_f32_16x16x32_bf16(false, a, false, b, (short)0, c,
                                                 false, false);
}

static __device__ __forceinline__ void storeOut(float* p, float v) { *p = v; }
static __device__ __forceinline__ void storeOut(unsigned short* p, float v) { *p = f2bf(v); }

// ---------------------------------------------------------------------------
// init: pack [phi_w; g_w] -> bf16 [256,256], stack biases, zero S accumulator
// grid 256x256 = 65536 threads
// ---------------------------------------------------------------------------
__global__ __launch_bounds__(256) void init_kernel(
    const float* __restrict__ phi_w, const float* __restrict__ phi_b,
    const float* __restrict__ g_w, const float* __restrict__ g_b,
    unsigned short* __restrict__ Wpg, float* __restrict__ pg_bias,
    float* __restrict__ S) {
  int t = blockIdx.x * 256 + threadIdx.x;              // 0 .. 65535
  Wpg[t] = f2bf(t < CI * CC ? phi_w[t] : g_w[t - CI * CC]);
  S[t] = 0.0f;                                         // 4*128*128 = 65536
  if (t < CI) pg_bias[t] = phi_b[t];
  else if (t < 2 * CI) pg_bias[t] = g_b[t - CI];
}

// ---------------------------------------------------------------------------
// gemm256<OutT>: Out[b, m, n] = sum_k A[b?, m, k](bf16) * x[b, k, n] + bias[b?, m]
// M = 256 (grid.y picks 128-row half), N tile = 64, K = 256.
// 256 threads = 8 wave32; wave w owns rows [16w,16w+16), 4 accum frags (N=64).
// ---------------------------------------------------------------------------
template <typename OutT>
__global__ __launch_bounds__(256) void gemm256_kernel(
    const unsigned short* __restrict__ A, long aStride,
    const float* __restrict__ bias, int biasStride,
    const float* __restrict__ X, OutT* __restrict__ Out) {
  __shared__ __align__(16) unsigned short ldsA[128 * 32];  // row-major [row][k]
  __shared__ __align__(16) unsigned short ldsX[64 * 32];   // col-major [col][k]

  const int tid = threadIdx.x;
  const int lane = tid & 31, wave = tid >> 5;
  const int hs = lane >> 4, l15 = lane & 15;
  const int b = blockIdx.x >> 6;
  const int n0 = (blockIdx.x & 63) * 64;
  const int mbase = blockIdx.y * 128;

  const unsigned short* Ab = A + (size_t)b * aStride;
  const float* Xb = X + (size_t)b * CN;
  const float* biasb = bias + (size_t)b * biasStride;

  const v8f z = {0, 0, 0, 0, 0, 0, 0, 0};
  v8f acc[4] = {z, z, z, z};

  // staging maps (loop-invariant)
  const int arow_ld = tid >> 1, aks = (tid & 1) * 16;          // A tile
  const int xcb = (tid & 15) * 4, xrp = (tid >> 4) * 2;        // X tile (2 rows x 4 cols)

  for (int k0 = 0; k0 < 256; k0 += 32) {
    {  // stage A tile [128 x 32] bf16 (already bf16 in global)
      const u16x8* src = (const u16x8*)(Ab + (size_t)(mbase + arow_ld) * 256 + k0 + aks);
      *(u16x8*)&ldsA[arow_ld * 32 + aks] = src[0];
      *(u16x8*)&ldsA[arow_ld * 32 + aks + 8] = src[1];
    }
    {  // stage X tile [32k x 64n] fp32 -> bf16x2 packed b32, column-major in LDS
      const f32x4* s0 = (const f32x4*)(Xb + (size_t)(k0 + xrp) * NN + n0 + xcb);
      const f32x4* s1 = (const f32x4*)(Xb + (size_t)(k0 + xrp + 1) * NN + n0 + xcb);
      f32x4 f0 = s0[0], f1 = s1[0];
#pragma unroll
      for (int i = 0; i < 4; ++i)
        *(unsigned int*)&ldsX[(xcb + i) * 32 + xrp] = f2bf2(f0[i], f1[i]);
    }
    __syncthreads();
    // A frag: lane l15 = row; lane-half selects K chunks {0-7,16-23}/{8-15,24-31}
    const unsigned short* arow = &ldsA[(wave * 16 + l15) * 32];
    v16bf a = mkfrag(*(const u16x8*)&arow[hs * 8], *(const u16x8*)&arow[16 + hs * 8]);
#pragma unroll
    for (int jf = 0; jf < 4; ++jf) {
      // B frag: lane l15 = column; lane-half selects K chunks {0-7,16-23}/{8-15,24-31}
      const unsigned short* bp = &ldsX[(jf * 16 + l15) * 32 + hs * 8];
      v16bf bb = mkfrag(*(const u16x8*)bp, *(const u16x8*)(bp + 16));
      acc[jf] = wmma_bf16(a, bb, acc[jf]);
    }
    __syncthreads();
  }
  OutT* outb = Out + (size_t)b * CN + (size_t)(mbase + wave * 16 + hs * 8) * NN + n0 + l15;
#pragma unroll
  for (int jf = 0; jf < 4; ++jf) {
#pragma unroll
    for (int r = 0; r < 8; ++r) {
      float v = acc[jf][r] + biasb[mbase + wave * 16 + hs * 8 + r];
      storeOut(outb + (size_t)r * NN + jf * 16, v);
    }
  }
}

// ---------------------------------------------------------------------------
// S[b,i,j] = sum_m phi[b,i,m] * g[b,j,m]   (K = N = 4096, split over grid.y)
// pg: [B][2*CI][N] bf16 (rows 0-127 = phi, 128-255 = g). fp32 atomic reduce.
// ---------------------------------------------------------------------------
__global__ __launch_bounds__(256) void s_kernel(const unsigned short* __restrict__ pg,
                                                float* __restrict__ S) {
  const int b = blockIdx.x, kz = blockIdx.y;
  const int tid = threadIdx.x, lane = tid & 31, wave = tid >> 5;
  const int hs = lane >> 4, l15 = lane & 15;
  const unsigned short* phi = pg + (size_t)b * (2 * CI * NN);
  const unsigned short* g = phi + (size_t)CI * NN;

  const v8f z = {0, 0, 0, 0, 0, 0, 0, 0};
  v8f acc[8] = {z, z, z, z, z, z, z, z};

  const unsigned short* ap = phi + (size_t)(wave * 16 + l15) * NN;
  const int kend = kz * 1024 + 1024;
  for (int k = kz * 1024; k < kend; k += 32) {
    v16bf a = mkfrag(*(const u16x8*)&ap[k + hs * 8], *(const u16x8*)&ap[k + 16 + hs * 8]);
#pragma unroll
    for (int jf = 0; jf < 8; ++jf) {
      const unsigned short* bp = g + (size_t)(jf * 16 + l15) * NN + k + hs * 8;
      v16bf bb = mkfrag(*(const u16x8*)bp, *(const u16x8*)(bp + 16));
      acc[jf] = wmma_bf16(a, bb, acc[jf]);
    }
  }
  float* Sb = S + (size_t)b * CI * CI;
#pragma unroll
  for (int jf = 0; jf < 8; ++jf)
#pragma unroll
    for (int r = 0; r < 8; ++r)
      atomicAdd(&Sb[(wave * 16 + r + hs * 8) * CI + jf * 16 + l15], acc[jf][r]);
}

// M1[b,i,c] = (1/N) * sum_j S[b,i,j] * w_w[c,j]      grid = B*128, threads = c
__global__ __launch_bounds__(256) void m1_kernel(const float* __restrict__ S,
                                                 const float* __restrict__ w_w,
                                                 float* __restrict__ M1) {
  int b = blockIdx.x >> 7, i = blockIdx.x & 127, c = threadIdx.x;
  const float* Sr = S + ((size_t)b * CI + i) * CI;
  const float* wr = w_w + (size_t)c * CI;
  float acc = 0.f;
#pragma unroll 4
  for (int j = 0; j < CI; ++j) acc += Sr[j] * wr[j];
  M1[((size_t)b * CI + i) * CC + c] = acc * (1.0f / NN);
}

// A2[b,c,c'] = sum_i theta_w[i,c'] * M1[b,i,c]  (bf16)   grid = B*256, thr = c'
__global__ __launch_bounds__(256) void a2_kernel(const float* __restrict__ theta_w,
                                                 const float* __restrict__ M1,
                                                 unsigned short* __restrict__ A2) {
  int b = blockIdx.x >> 8, c = blockIdx.x & 255, cp = threadIdx.x;
  const float* Mb = M1 + (size_t)b * CI * CC;
  float acc = 0.f;
#pragma unroll 4
  for (int i = 0; i < CI; ++i)
    acc += theta_w[(size_t)i * CC + cp] * Mb[(size_t)i * CC + c];
  A2[((size_t)b * CC + c) * CC + cp] = f2bf(acc);
}

// q[b,c] = w_b[c] + sum_i theta_b[i] * M1[b,i,c]        grid = B, thr = c
__global__ __launch_bounds__(256) void q_kernel(const float* __restrict__ theta_b,
                                                const float* __restrict__ M1,
                                                const float* __restrict__ w_b,
                                                float* __restrict__ qv) {
  int b = blockIdx.x, c = threadIdx.x;
  const float* Mb = M1 + (size_t)b * CI * CC;
  float acc = w_b[c];
#pragma unroll 4
  for (int i = 0; i < CI; ++i) acc += theta_b[i] * Mb[(size_t)i * CC + c];
  qv[b * CC + c] = acc;
}

// per-channel mean / invstd over (B, N)                  grid = 256 channels
__global__ __launch_bounds__(256) void stats_kernel(const float* __restrict__ wy,
                                                    float* __restrict__ stats) {
  __shared__ float ls[256], lq[256];
  int c = blockIdx.x, t = threadIdx.x;
  float s = 0.f, q = 0.f;
  for (int b = 0; b < BB; ++b) {
    const float* p = wy + (size_t)b * CN + (size_t)c * NN;
    for (int n = t; n < NN; n += 256) {
      float v = p[n];
      s += v;
      q += v * v;
    }
  }
  ls[t] = s;
  lq[t] = q;
  __syncthreads();
  for (int off = 128; off > 0; off >>= 1) {
    if (t < off) { ls[t] += ls[t + off]; lq[t] += lq[t + off]; }
    __syncthreads();
  }
  if (t == 0) {
    float mean = ls[0] * (1.0f / (BB * NN));
    float var = lq[0] * (1.0f / (BB * NN)) - mean * mean;
    stats[2 * c] = mean;
    stats[2 * c + 1] = rsqrtf(var + 1e-5f);
  }
}

// out = (wy - mean) * invstd * gamma + beta + x          float4 vectorized
__global__ __launch_bounds__(256) void final_kernel(
    const float* __restrict__ wy, const float* __restrict__ x,
    const float* __restrict__ stats, const float* __restrict__ gamma,
    const float* __restrict__ beta, float* __restrict__ out) {
  int idx = blockIdx.x * 256 + threadIdx.x;  // 1048576 float4's total
  int c = (idx >> 10) & 255;                 // NN/4 = 1024 float4 per channel row
  f32x4 w = ((const f32x4*)wy)[idx];
  f32x4 xv = ((const f32x4*)x)[idx];
  float mean = stats[2 * c], inv = stats[2 * c + 1];
  float gm = gamma[c], bt = beta[c];
  f32x4 o;
#pragma unroll
  for (int i = 0; i < 4; ++i) o[i] = (w[i] - mean) * inv * gm + bt + xv[i];
  ((f32x4*)out)[idx] = o;
}

// ---------------------------------------------------------------------------
extern "C" void kernel_launch(void* const* d_in, const int* in_sizes, int n_in,
                              void* d_out, int out_size, void* d_ws, size_t ws_size,
                              hipStream_t stream) {
  const float* x       = (const float*)d_in[0];
  const float* theta_w = (const float*)d_in[1];
  const float* theta_b = (const float*)d_in[2];
  const float* phi_w   = (const float*)d_in[3];
  const float* phi_b   = (const float*)d_in[4];
  const float* g_w     = (const float*)d_in[5];
  const float* g_b     = (const float*)d_in[6];
  const float* w_w     = (const float*)d_in[7];
  const float* w_b     = (const float*)d_in[8];
  const float* gamma   = (const float*)d_in[9];
  const float* beta    = (const float*)d_in[10];
  (void)in_sizes; (void)n_in; (void)out_size; (void)ws_size;

  char* ws = (char*)d_ws;                                      // bytes
  unsigned short* pg  = (unsigned short*)(ws);                 //  8,388,608  [B][256][N] bf16
  float* S            = (float*)(ws + 8388608);                //    262,144  [B][128][128]
  float* M1           = (float*)(ws + 8650752);                //    524,288  [B][128][256]
  unsigned short* A2  = (unsigned short*)(ws + 9175040);       //    524,288  [B][256][256] bf16
  float* qv           = (float*)(ws + 9699328);                //      4,096  [B][256]
  float* wy           = (float*)(ws + 9703424);                // 16,777,216  [B][256][N]
  float* stats        = (float*)(ws + 26480640);               //      2,048  [256][2]
  unsigned short* Wpg = (unsigned short*)(ws + 26482688);      //    131,072  [256][256] bf16
  float* pg_bias      = (float*)(ws + 26613760);               //      1,024  [256]

  init_kernel<<<256, 256, 0, stream>>>(phi_w, phi_b, g_w, g_b, Wpg, pg_bias, S);

  dim3 gg(BB * 64, 2);  // 64 N-tiles per batch, 2 M-halves
  // pg = [phi_w; g_w] * x + bias   (bf16 out)
  gemm256_kernel<unsigned short><<<gg, 256, 0, stream>>>(Wpg, 0L, pg_bias, 0, x, pg);

  dim3 gs(BB, 4);       // split-K over N
  s_kernel<<<gs, 256, 0, stream>>>(pg, S);

  m1_kernel<<<BB * 128, 256, 0, stream>>>(S, w_w, M1);
  a2_kernel<<<BB * 256, 256, 0, stream>>>(theta_w, M1, A2);
  q_kernel<<<BB, 256, 0, stream>>>(theta_b, M1, w_b, qv);

  // wy = A2 * x + q   (fp32 out)
  gemm256_kernel<float><<<gg, 256, 0, stream>>>(A2, 65536L, qv, 256, x, wy);

  stats_kernel<<<256, 256, 0, stream>>>(wy, stats);
  final_kernel<<<4096, 256, 0, stream>>>(wy, x, stats, gamma, beta, (float*)d_out);
}
